// QueryAssembly_10453950399145
// MI455X (gfx1250) — matrix operand order repaired and checked by reading, not compiled, not run on hardware
//
#include <hip/hip_runtime.h>
#include <math.h>

typedef _Float16 v16h __attribute__((ext_vector_type(16)));
typedef _Float16 v8h  __attribute__((ext_vector_type(8)));
typedef float    v8f  __attribute__((ext_vector_type(8)));

#define B_  32
#define L_  512
#define D_  768
#define K_  8192
#define INF_F __builtin_inff()

// LDS byte budgets
#define RS_QBYTES  (64 * D_ * 2)            // 96 KB  Q block (rowstats)
#define RS_PTILE   (32 * D_ * 2)            // 48 KB  P tile (32 columns)
#define RS_SMEM    (RS_QBYTES + 2 * RS_PTILE)          // 192 KB
#define CM_PBYTES  (128 * D_ * 2)           // 192 KB P block (colmax)
#define CM_QTILE   (32 * D_ * 2)            // 48 KB  Q tile (32 rows)
#define CM_SMEM    (CM_PBYTES + 2 * CM_QTILE)          // 288 KB

// ---- CDNA5 async copy global->LDS (GLOBAL_LOAD_ASYNC_TO_LDS_B128, ASYNCcnt)
__device__ __forceinline__ void async_b128(void* lds_dst, const void* gsrc) {
  unsigned l = (unsigned)(uintptr_t)lds_dst;                 // LDS byte address
  unsigned long long g = (unsigned long long)(uintptr_t)gsrc;
  asm volatile("global_load_async_to_lds_b128 %0, %1, off"
               :: "v"(l), "v"(g) : "memory");
}
__device__ __forceinline__ void stage_async(char* lds_dst, const char* gsrc,
                                            int bytes, int tid, int nthr) {
  for (int ofs = tid * 16; ofs < bytes; ofs += nthr * 16)
    async_b128(lds_dst + ofs, gsrc + ofs);
}
__device__ __forceinline__ void wait_async0() {
  asm volatile("s_wait_asynccnt 0" ::: "memory");
}

__device__ __forceinline__ v16h load_a(const _Float16* qrow, int dk, int kbase) {
  v8h alo = *(const v8h*)(qrow + dk + 2 * kbase);
  v8h ahi = *(const v8h*)(qrow + dk + 2 * kbase + 8);
  return __builtin_shufflevector(alo, ahi,
           0,1,2,3,4,5,6,7,8,9,10,11,12,13,14,15);
}
#define WMMA(A, Bm, Cm) __builtin_amdgcn_wmma_f32_16x16x32_f16( \
                          false, (A), false, (Bm), (short)0, (Cm), false, false)

// ---------------- Kernel 0: f32 -> f16 conversion + zero per-batch sum-sq ----
__global__ void k_convert(const float* __restrict__ q,
                          const float* __restrict__ p,
                          _Float16* __restrict__ qh,
                          _Float16* __restrict__ ph,
                          float* __restrict__ ss) {
  long long idx = (long long)blockIdx.x * blockDim.x + threadIdx.x;
  const long long QN = (long long)B_ * L_ * D_;
  const long long PN = (long long)K_ * D_;
  if (idx < QN)            qh[idx]      = (_Float16)q[idx];
  else if (idx < QN + PN)  ph[idx - QN] = (_Float16)p[idx - QN];
  if (idx < B_) ss[idx] = 0.0f;
}

// ---------------- Kernel 1: per-row softmax stats c[b,l] = max + log(sumexp) -
// Block = 4 waves = 64 complete rows, Q block LDS-resident. 32-column P tiles
// double-buffered via async copies; each wave runs TWO accumulators sharing
// one A fragment, with software-pipelined LDS fragment loads.
__global__ void k_rowstats(const _Float16* __restrict__ qh,
                           const _Float16* __restrict__ ph,
                           const float* __restrict__ temp,
                           const int* __restrict__ mask,
                           float* __restrict__ C) {
  extern __shared__ __align__(32) char smem[];
  _Float16* Qs = (_Float16*)smem;
  char* Pb0 = smem + RS_QBYTES;
  char* Pb1 = smem + RS_QBYTES + RS_PTILE;

  const int rb     = blockIdx.x;           // 64-row block over B*L
  const int tid    = threadIdx.x;
  const int wave   = tid >> 5;
  const int lane   = tid & 31;
  const int half   = lane >> 4;
  const int lane15 = lane & 15;
  const int kbase  = half * 8;
  const float invT = 1.0f / temp[0];

  stage_async((char*)Qs, (const char*)(qh + (long long)rb * 64 * D_),
              RS_QBYTES, tid, 128);
  stage_async(Pb0, (const char*)ph, RS_PTILE, tid, 128);
  wait_async0();
  __syncthreads();

  const _Float16* qrow = Qs + (wave * 16 + lane15) * D_;   // A row (M=lane15)

  float mrun[8], srun[8];
  #pragma unroll
  for (int v = 0; v < 8; ++v) { mrun[v] = -INF_F; srun[v] = 0.0f; }

  for (int nt = 0; nt < K_ / 32; ++nt) {                   // 32-col P tiles
    if (nt + 1 < K_ / 32)                                  // prefetch next P
      stage_async(((nt + 1) & 1) ? Pb1 : Pb0,
                  (const char*)(ph + (long long)(nt + 1) * 32 * D_),
                  RS_PTILE, tid, 128);

    const _Float16* pb = (const _Float16*)((nt & 1) ? Pb1 : Pb0);
    const _Float16* prow0 = pb + lane15 * D_ + half * 16;  // B cols n..n+15
    const _Float16* prow1 = prow0 + 16 * D_;               // B cols n+16..n+31

    v8f acc0 = {}, acc1 = {};
    v16h a  = load_a(qrow, 0, kbase);                      // pipelined frags
    v16h b0 = *(const v16h*)(prow0);
    v16h b1 = *(const v16h*)(prow1);
    #pragma unroll 4
    for (int dk = 0; dk < D_ - 32; dk += 32) {
      v16h an  = load_a(qrow, dk + 32, kbase);
      v16h b0n = *(const v16h*)(prow0 + dk + 32);
      v16h b1n = *(const v16h*)(prow1 + dk + 32);
      acc0 = WMMA(a, b0, acc0);
      acc1 = WMMA(a, b1, acc1);
      a = an; b0 = b0n; b1 = b1n;
    }
    acc0 = WMMA(a, b0, acc0);
    acc1 = WMMA(a, b1, acc1);

    #pragma unroll
    for (int v = 0; v < 8; ++v) {          // online softmax, branchless
      float x0 = acc0[v] * invT;
      float nm = fmaxf(mrun[v], x0);
      srun[v]  = srun[v] * __expf(mrun[v] - nm) + __expf(x0 - nm);
      mrun[v]  = nm;
      float x1 = acc1[v] * invT;
      nm       = fmaxf(mrun[v], x1);
      srun[v]  = srun[v] * __expf(mrun[v] - nm) + __expf(x1 - nm);
      mrun[v]  = nm;
    }
    wait_async0();
    __syncthreads();
  }

  // reduce (max,sum) across the 16 lanes of each half (different N columns)
  #pragma unroll
  for (int off = 1; off < 16; off <<= 1) {
    #pragma unroll
    for (int v = 0; v < 8; ++v) {
      float om = __shfl_xor(mrun[v], off, 32);
      float os = __shfl_xor(srun[v], off, 32);
      float nm = fmaxf(mrun[v], om);
      srun[v]  = srun[v] * __expf(mrun[v] - nm) + os * __expf(om - nm);
      mrun[v]  = nm;
    }
  }
  if (lane15 == 0) {                       // C-slot v, half h -> row v + 8h
    #pragma unroll
    for (int v = 0; v < 8; ++v) {
      int row = rb * 64 + wave * 16 + v + 8 * half;
      float c = mrun[v] + __logf(srun[v]);
      C[row] = (mask[row] != 0) ? c : INF_F;   // +inf kills masked rows
    }
  }
}

// ---------------- Kernel 2: s[b,k] = exp(max_l(logit - c_l)); Σ s^2 per batch
// Block = 8 waves = 128 columns; 128x768 P block LDS-resident. 32-row Q tiles
// double-buffered; each wave runs TWO accumulators (two M-tiles) sharing one
// B fragment, with software-pipelined LDS fragment loads.
__global__ void k_colmax(const _Float16* __restrict__ qh,
                         const _Float16* __restrict__ ph,
                         const float* __restrict__ temp,
                         const float* __restrict__ C,
                         float* __restrict__ out,
                         float* __restrict__ ss) {
  extern __shared__ __align__(32) char smem[];
  _Float16* Ps = (_Float16*)smem;
  char* Qb0 = smem + CM_PBYTES;
  char* Qb1 = smem + CM_PBYTES + CM_QTILE;

  const int b      = blockIdx.y;
  const int tid    = threadIdx.x;
  const int wave   = tid >> 5;
  const int lane   = tid & 31;
  const int half   = lane >> 4;
  const int lane15 = lane & 15;
  const int kbase  = half * 8;
  const int n0     = blockIdx.x * 128 + wave * 16;
  const float invT = 1.0f / temp[0];

  stage_async((char*)Ps, (const char*)(ph + (long long)blockIdx.x * 128 * D_),
              CM_PBYTES, tid, 256);
  stage_async(Qb0, (const char*)(qh + (long long)b * L_ * D_),
              CM_QTILE, tid, 256);
  wait_async0();
  __syncthreads();

  const _Float16* prow = Ps + (wave * 16 + lane15) * D_ + half * 16;  // B col

  float smax = 0.0f;

  for (int lt = 0; lt < L_ / 32; ++lt) {                   // 32-row Q tiles
    if (lt + 1 < L_ / 32)                                  // prefetch next Q
      stage_async(((lt + 1) & 1) ? Qb1 : Qb0,
                  (const char*)(qh + ((long long)b * L_ + (lt + 1) * 32) * D_),
                  CM_QTILE, tid, 256);

    const _Float16* qb = (const _Float16*)((lt & 1) ? Qb1 : Qb0);
    const _Float16* qrow0 = qb + lane15 * D_;              // rows l..l+15
    const _Float16* qrow1 = qb + (16 + lane15) * D_;       // rows l+16..l+31

    v8f acc0 = {}, acc1 = {};
    v16h a0 = load_a(qrow0, 0, kbase);                     // pipelined frags
    v16h a1 = load_a(qrow1, 0, kbase);
    v16h bf = *(const v16h*)(prow);
    #pragma unroll 4
    for (int dk = 0; dk < D_ - 32; dk += 32) {
      v16h a0n = load_a(qrow0, dk + 32, kbase);
      v16h a1n = load_a(qrow1, dk + 32, kbase);
      v16h bfn = *(const v16h*)(prow + dk + 32);
      acc0 = WMMA(a0, bf, acc0);
      acc1 = WMMA(a1, bf, acc1);
      a0 = a0n; a1 = a1n; bf = bfn;
    }
    acc0 = WMMA(a0, bf, acc0);
    acc1 = WMMA(a1, bf, acc1);

    const float* crow0 = C + b * L_ + lt * 32 + 8 * half;
    const float* crow1 = crow0 + 16;
    #pragma unroll
    for (int v = 0; v < 8; ++v) {
      float x0 = acc0[v] * invT;
      smax = fmaxf(smax, __expf(x0 - crow0[v]));  // exp(x - inf)=0 if masked
      float x1 = acc1[v] * invT;
      smax = fmaxf(smax, __expf(x1 - crow1[v]));
    }
    wait_async0();
    __syncthreads();
  }

  // combine lane L with L+16 (same N column, other half of M rows)
  smax = fmaxf(smax, __shfl_xor(smax, 16, 32));

  float p = (lane < 16) ? smax * smax : 0.0f;    // partial Σ s^2 (no dupes)
  #pragma unroll
  for (int off = 1; off < 32; off <<= 1) p += __shfl_xor(p, off, 32);

  if (lane < 16) out[(long long)b * K_ + n0 + lane15] = smax;
  if (lane == 0) atomicAdd(&ss[b], p);
}

// ---------------- Kernel 3: L2 normalize over K -----------------------------
__global__ void k_norm(float* __restrict__ out, const float* __restrict__ ss) {
  int idx = blockIdx.x * blockDim.x + threadIdx.x;
  int b = idx / K_;
  float n = fmaxf(sqrtf(ss[b]), 1e-12f);
  out[idx] = out[idx] / n;
}

extern "C" void kernel_launch(void* const* d_in, const int* in_sizes, int n_in,
                              void* d_out, int out_size, void* d_ws, size_t ws_size,
                              hipStream_t stream) {
  (void)in_sizes; (void)n_in; (void)out_size; (void)ws_size;
  const float* q    = (const float*)d_in[0];   // (B,L,d) f32
  const float* p    = (const float*)d_in[1];   // (K,d)   f32
  const float* temp = (const float*)d_in[2];   // scalar  f32
  const int*   mask = (const int*)d_in[3];     // (B,L)   i32
  float* out = (float*)d_out;                  // (B,K)   f32

  const long long QN = (long long)B_ * L_ * D_;
  const long long PN = (long long)K_ * D_;

  _Float16* qh = (_Float16*)d_ws;              // 25.2 MB
  _Float16* ph = qh + QN;                      // 12.6 MB
  float*    C  = (float*)(ph + PN);            // 64 KB  (c[b,l], +inf if masked)
  float*    ss = C + B_ * L_;                  // 128 B  (Σ s^2 per batch)

  const long long NCONV = QN + PN;
  k_convert<<<(unsigned)((NCONV + 255) / 256), 256, 0, stream>>>(q, p, qh, ph, ss);
  k_rowstats<<<(B_ * L_) / 64, 128, RS_SMEM, stream>>>(qh, ph, temp, mask, C);
  dim3 g2(K_ / 128, B_);
  k_colmax<<<g2, 256, CM_SMEM, stream>>>(qh, ph, temp, C, out, ss);
  k_norm<<<(B_ * K_) / 256, 256, 0, stream>>>(out, ss);
}
